// ONNXLanguageTransformer_59485297050057
// MI455X (gfx1250) — matrix operand order repaired and checked by reading, not compiled, not run on hardware
//
#include <hip/hip_runtime.h>
#include <math.h>

// ---------------------------------------------------------------------------
// Transformer forward for MI455X (gfx1250, wave32, WMMA).
// All GEMM-shaped math on v_wmma_f32_16x16x32_bf16 (bf16 in, f32 acc).
// Fragment feeds are b128 LDS/global loads matching ISA 7.12.2 layouts.
// GEMM uses double-buffered LDS staging (one barrier / k-step, HBM latency
// hidden behind WMMA). LayerNorm / softmax / residuals stay in fp32.
// ---------------------------------------------------------------------------

#define D_MODEL  256
#define NHEAD    8
#define HEAD_DIM 32
#define FF_DIM   1024
#define VOCAB    233
#define S_SRC    512
#define S_TGT    256
#define BATCH    32
#define LN_EPS   1e-5f

typedef __attribute__((ext_vector_type(16))) __bf16 v16bf;
typedef __attribute__((ext_vector_type(8)))  float  v8f;

union BF16Frag {
  v16bf v;
  unsigned short u[16];
  uint4 q[2];      // two 16-byte chunks -> b128 loads
};

union U4H8 { uint4 v; unsigned short h[8]; };

__device__ __forceinline__ unsigned short f2bf(float f) {
  unsigned int u = __float_as_uint(f);
  u += 0x7FFFu + ((u >> 16) & 1u);   // round-to-nearest-even
  return (unsigned short)(u >> 16);
}
__device__ __forceinline__ unsigned int pack2bf(float a, float b) {
  return (unsigned int)f2bf(a) | ((unsigned int)f2bf(b) << 16);
}

// --------------------------- elementwise kernels ---------------------------

__global__ void k_convert_bf16(const float* __restrict__ in,
                               unsigned short* __restrict__ out, int n4) {
  int i = blockIdx.x * blockDim.x + threadIdx.x;
  if (i >= n4) return;
  float4 f = ((const float4*)in)[i];
  uint2 o;
  o.x = pack2bf(f.x, f.y);
  o.y = pack2bf(f.z, f.w);
  ((uint2*)out)[i] = o;
}

__global__ void k_embed_src(const float* __restrict__ src, const float* __restrict__ pe,
                            float* __restrict__ xf, unsigned short* __restrict__ xbf,
                            int total2) {   // pairs
  int i = blockIdx.x * blockDim.x + threadIdx.x;
  if (i >= total2) return;
  int e = i * 2;
  int s = e / (BATCH * D_MODEL);
  int d = e % D_MODEL;
  float2 sv = ((const float2*)src)[i];
  float v0 = sv.x * 16.0f + pe[s * D_MODEL + d];       // sqrt(256) = 16
  float v1 = sv.y * 16.0f + pe[s * D_MODEL + d + 1];
  ((float2*)xf)[i] = make_float2(v0, v1);
  ((unsigned int*)xbf)[i] = pack2bf(v0, v1);
}

__global__ void k_embed_tgt(const int* __restrict__ tgt, const float* __restrict__ emb,
                            const float* __restrict__ pe,
                            float* __restrict__ xf, unsigned short* __restrict__ xbf,
                            int total2) {
  int i = blockIdx.x * blockDim.x + threadIdx.x;
  if (i >= total2) return;
  int e = i * 2;
  int s = e / (BATCH * D_MODEL);
  int rem = e % (BATCH * D_MODEL);
  int b = rem / D_MODEL;
  int d = rem % D_MODEL;
  int tok = tgt[s * BATCH + b];
  float2 ev = ((const float2*)(emb + (size_t)tok * D_MODEL))[d >> 1];
  float v0 = ev.x * 16.0f + pe[s * D_MODEL + d];
  float v1 = ev.y * 16.0f + pe[s * D_MODEL + d + 1];
  ((float2*)xf)[i] = make_float2(v0, v1);
  ((unsigned int*)xbf)[i] = pack2bf(v0, v1);
}

// (S,B,D) f32 -> (B*H, S, hd) bf16   (pairs along hd)
__global__ void k_split_heads(const float* __restrict__ x, unsigned short* __restrict__ o,
                              int S, int total2) {
  int i = blockIdx.x * blockDim.x + threadIdx.x;
  if (i >= total2) return;
  int e = i * 2;
  int bh = e / (S * HEAD_DIM);
  int rem = e % (S * HEAD_DIM);
  int s = rem / HEAD_DIM;
  int d = rem % HEAD_DIM;
  int b = bh / NHEAD, h = bh % NHEAD;
  const float* p = x + ((size_t)s * BATCH + b) * D_MODEL + h * HEAD_DIM + d;
  ((unsigned int*)o)[i] = pack2bf(p[0], p[1]);
}

// (S,B,D) f32 -> (B*H, hd, S) bf16  (head-transposed, pairs along S)
__global__ void k_split_heads_T(const float* __restrict__ x, unsigned short* __restrict__ o,
                                int S, int total2) {
  int i = blockIdx.x * blockDim.x + threadIdx.x;
  if (i >= total2) return;
  int e = i * 2;
  int bh = e / (HEAD_DIM * S);
  int rem = e % (HEAD_DIM * S);
  int d = rem / S;
  int s = rem % S;
  int b = bh / NHEAD, h = bh % NHEAD;
  float v0 = x[((size_t)s * BATCH + b) * D_MODEL + h * HEAD_DIM + d];
  float v1 = x[((size_t)(s + 1) * BATCH + b) * D_MODEL + h * HEAD_DIM + d];
  ((unsigned int*)o)[i] = pack2bf(v0, v1);
}

// (B*H, S, hd) f32 -> (S,B,D) bf16  (pairs along D)
__global__ void k_merge_heads(const float* __restrict__ oh, unsigned short* __restrict__ o,
                              int S, int total2) {
  int i = blockIdx.x * blockDim.x + threadIdx.x;
  if (i >= total2) return;
  int e = i * 2;
  int s = e / (BATCH * D_MODEL);
  int rem = e % (BATCH * D_MODEL);
  int b = rem / D_MODEL;
  int dd = rem % D_MODEL;
  int h = dd / HEAD_DIM, d = dd % HEAD_DIM;
  const float* p = oh + (((size_t)b * NHEAD + h) * S + s) * HEAD_DIM + d;
  ((unsigned int*)o)[i] = pack2bf(p[0], p[1]);
}

// Fused residual-add + LayerNorm. One wave per token, D=256 (8 elems/lane).
__global__ __launch_bounds__(32)
void k_ln(const float* __restrict__ x, const float* __restrict__ y,
          const float* __restrict__ g, const float* __restrict__ b,
          float* __restrict__ of, unsigned short* __restrict__ obf) {
  const int t = blockIdx.x;
  const int lane = threadIdx.x;
  float v[8];
  float s = 0.f;
#pragma unroll
  for (int i = 0; i < 8; ++i) {
    int d = lane + i * 32;
    float val = x[(size_t)t * D_MODEL + d];
    if (y) val += y[(size_t)t * D_MODEL + d];
    v[i] = val;
    s += val;
  }
#pragma unroll
  for (int o = 16; o > 0; o >>= 1) s += __shfl_xor(s, o, 32);
  float mu = s * (1.0f / D_MODEL);
  float var = 0.f;
#pragma unroll
  for (int i = 0; i < 8; ++i) { float d2 = v[i] - mu; var += d2 * d2; }
#pragma unroll
  for (int o = 16; o > 0; o >>= 1) var += __shfl_xor(var, o, 32);
  var *= (1.0f / D_MODEL);
  float r = rsqrtf(var + LN_EPS);
#pragma unroll
  for (int i = 0; i < 8; ++i) {
    int d = lane + i * 32;
    float ov = (v[i] - mu) * r * g[d] + b[d];
    if (of)  of[(size_t)t * D_MODEL + d] = ov;
    if (obf) obf[(size_t)t * D_MODEL + d] = f2bf(ov);
  }
}

// ------------------------------- GEMM (WMMA) -------------------------------
// C[M,N] = A[M,K](bf16) x B[K,N](bf16) + bias, optional relu, optional bf16
// copy-out, optional row permutation (s*B+b -> b*S+s) for the final FC.
// 8 waves/block; block tile 64x128; wave tile 32x32 (2x2 WMMA); K in 32-wide
// LDS tiles. A row-major, B transposed in LDS (lBt[n][k]) so every fragment
// is two ds_load_b128. VECB=true: full-tile vector staging, double-buffered
// LDS (one barrier per k-step, HBM latency hidden). VECB=false: bounds-
// checked scalar staging (final FC, N=233, only).
// M multiple of 64, K multiple of 32 at all call sites.

__device__ __forceinline__ void gemm_frag_mma(
    const unsigned short* lAp, const unsigned short* lBtp,
    int waveM, int waveN, int khalf, int l16, v8f acc[2][2]) {
  BF16Frag fa[2], fb[2];
#pragma unroll
  for (int mt = 0; mt < 2; ++mt) {
    int m = waveM * 32 + mt * 16 + l16;
    fa[mt].q[0] = *(const uint4*)&lAp[m * 32 + khalf * 8];
    fa[mt].q[1] = *(const uint4*)&lAp[m * 32 + 16 + khalf * 8];
  }
#pragma unroll
  for (int nt = 0; nt < 2; ++nt) {
    int n = waveN * 32 + nt * 16 + l16;
    fb[nt].q[0] = *(const uint4*)&lBtp[n * 32 + khalf * 8];
    fb[nt].q[1] = *(const uint4*)&lBtp[n * 32 + 16 + khalf * 8];
  }
#pragma unroll
  for (int mt = 0; mt < 2; ++mt)
#pragma unroll
    for (int nt = 0; nt < 2; ++nt)
      acc[mt][nt] = __builtin_amdgcn_wmma_f32_16x16x32_bf16(
          false, fa[mt].v, false, fb[nt].v, (short)0, acc[mt][nt], false, false);
}

template <bool VECB>
__global__ __launch_bounds__(256)
void k_gemm_bf16(const unsigned short* __restrict__ A,
                 const unsigned short* __restrict__ Bw,
                 const float* __restrict__ bias,
                 float* __restrict__ C,
                 unsigned short* __restrict__ Cbf,
                 int M, int N, int K, int relu, int permB) {
  const int tid = threadIdx.x;
  const int lane = tid & 31;
  const int wid = tid >> 5;
  const int waveM = wid >> 2;   // 0..1
  const int waveN = wid & 3;    // 0..3
  const int mBase = blockIdx.y * 64;
  const int nBase = blockIdx.x * 128;
  const int khalf = lane >> 4;
  const int l16 = lane & 15;
  // staging indices
  const int ar = tid >> 2;               // 0..63
  const int ac = (tid & 3) * 8;          // 0,8,16,24
  const int bk = (tid >> 4) * 2;         // 0,2,..,30
  const int bn = (tid & 15) * 8;         // 0..120

  v8f acc[2][2] = {};

  if constexpr (VECB) {
    __shared__ unsigned short lA[2][64 * 32];
    __shared__ unsigned short lBt[2][128 * 32];
    const int nSteps = K >> 5;

    // prologue: stage tile 0
    uint4 ra = *(const uint4*)&A[(size_t)(mBase + ar) * K + ac];
    U4H8 rb0, rb1;
    rb0.v = *(const uint4*)&Bw[(size_t)bk * N + (nBase + bn)];
    rb1.v = *(const uint4*)&Bw[(size_t)(bk + 1) * N + (nBase + bn)];
    *(uint4*)&lA[0][ar * 32 + ac] = ra;
#pragma unroll
    for (int i = 0; i < 8; ++i)
      *(unsigned int*)&lBt[0][(bn + i) * 32 + bk] =
          (unsigned int)rb0.h[i] | ((unsigned int)rb1.h[i] << 16);
    __syncthreads();

    for (int t = 0; t < nSteps; ++t) {
      const int cur = t & 1, nxt = cur ^ 1;
      const bool more = (t + 1) < nSteps;
      if (more) {   // prefetch next tile into registers while computing
        int k0 = (t + 1) * 32;
        ra = *(const uint4*)&A[(size_t)(mBase + ar) * K + k0 + ac];
        rb0.v = *(const uint4*)&Bw[(size_t)(k0 + bk) * N + (nBase + bn)];
        rb1.v = *(const uint4*)&Bw[(size_t)(k0 + bk + 1) * N + (nBase + bn)];
      }
      gemm_frag_mma(lA[cur], lBt[cur], waveM, waveN, khalf, l16, acc);
      if (more) {   // fill the idle buffer (last read was t-1, barrier since)
        *(uint4*)&lA[nxt][ar * 32 + ac] = ra;
#pragma unroll
        for (int i = 0; i < 8; ++i)
          *(unsigned int*)&lBt[nxt][(bn + i) * 32 + bk] =
              (unsigned int)rb0.h[i] | ((unsigned int)rb1.h[i] << 16);
      }
      __syncthreads();
    }
  } else {
    __shared__ unsigned short lA1[64 * 32];
    __shared__ unsigned short lBt1[128 * 32];
    for (int k0 = 0; k0 < K; k0 += 32) {
      __syncthreads();
      *(uint4*)&lA1[ar * 32 + ac] = *(const uint4*)&A[(size_t)(mBase + ar) * K + k0 + ac];
#pragma unroll
      for (int i = 0; i < 16; ++i) {
        int e = tid + i * 256;          // 4096 elems
        int r = e >> 7, c = e & 127;    // k, n
        int n = nBase + c;
        lBt1[c * 32 + r] = (n < N) ? Bw[(size_t)(k0 + r) * N + n] : (unsigned short)0;
      }
      __syncthreads();
      gemm_frag_mma(lA1, lBt1, waveM, waveN, khalf, l16, acc);
    }
  }

  // epilogue: bias + relu + store (f32, optional bf16)
#pragma unroll
  for (int mt = 0; mt < 2; ++mt) {
#pragma unroll
    for (int nt = 0; nt < 2; ++nt) {
#pragma unroll
      for (int r = 0; r < 8; ++r) {
        int gm = mBase + waveM * 32 + mt * 16 + khalf * 8 + r;
        int gn = nBase + waveN * 32 + nt * 16 + l16;
        if (gn < N) {
          float val = acc[mt][nt][r];
          if (bias) val += bias[gn];
          if (relu && val < 0.f) val = 0.f;
          int row = gm;
          if (permB > 0) {
            int Sr = M / permB;
            row = (gm % permB) * Sr + (gm / permB);
          }
          C[(size_t)row * N + gn] = val;
          if (Cbf) Cbf[(size_t)row * N + gn] = f2bf(val);
        }
      }
    }
  }
}

// ------------------------------ Attention ---------------------------------
// One wave per (batch*head, 16-query tile). K dim = head_dim = 32, so Q*K^T
// is one WMMA per 16-key tile. Q/K fragments: 2x global b128 each (row-major
// per head). V comes head-transposed (B*H, hd, S) so PV B-fragments are also
// 2x b128. Scores (16 x Sk) live in LDS; softmax split across all 32 lanes.
__global__ __launch_bounds__(32)
void k_attn(const unsigned short* __restrict__ Qh,
            const unsigned short* __restrict__ Kh,
            const unsigned short* __restrict__ Vt,
            float* __restrict__ Oh,
            int Sq, int Sk, int causal, float scale) {
  __shared__ float sc[16 * 512];
  __shared__ unsigned short wb[16 * 512];
  const int lane = threadIdx.x;
  const int khalf = lane >> 4;
  const int l16 = lane & 15;
  const int qTiles = Sq >> 4;
  const int bh = blockIdx.x / qTiles;
  const int qt = blockIdx.x % qTiles;

  // Q fragment (A 16x32): two contiguous 8-half runs per lane
  BF16Frag qa;
  {
    const unsigned short* qp = Qh + ((size_t)bh * Sq + qt * 16 + l16) * HEAD_DIM;
    qa.q[0] = *(const uint4*)(qp + khalf * 8);
    qa.q[1] = *(const uint4*)(qp + 16 + khalf * 8);
  }

  // scores = Q K^T * scale (+ causal mask)
  for (int kt = 0; kt < (Sk >> 4); ++kt) {
    BF16Frag kb;
    {
      const unsigned short* kp = Kh + ((size_t)bh * Sk + kt * 16 + l16) * HEAD_DIM;
      kb.q[0] = *(const uint4*)(kp + khalf * 8);
      kb.q[1] = *(const uint4*)(kp + 16 + khalf * 8);
    }
    v8f z = {};
    v8f c = __builtin_amdgcn_wmma_f32_16x16x32_bf16(
        false, qa.v, false, kb.v, (short)0, z, false, false);
#pragma unroll
    for (int r = 0; r < 8; ++r) {
      int m = khalf * 8 + r;
      int key = kt * 16 + l16;
      float v = c[r] * scale;
      if (causal && key > qt * 16 + m) v = -1e30f;
      sc[m * 512 + key] = v;
    }
  }
  __syncthreads();

  // row softmax -> bf16 weights in LDS (2 lanes per row)
  {
    int row = l16;
    int cnt = Sk >> 1;
    int kbeg = khalf * cnt;
    float mx = -1e30f;
    for (int k = 0; k < cnt; ++k) mx = fmaxf(mx, sc[row * 512 + kbeg + k]);
    mx = fmaxf(mx, __shfl_xor(mx, 16, 32));
    float sum = 0.f;
    for (int k = 0; k < cnt; ++k) {
      float e = __expf(sc[row * 512 + kbeg + k] - mx);
      sc[row * 512 + kbeg + k] = e;
      sum += e;
    }
    sum += __shfl_xor(sum, 16, 32);
    float inv = 1.f / sum;
    for (int k = 0; k < cnt; k += 2) {
      *(unsigned int*)&wb[row * 512 + kbeg + k] =
          pack2bf(sc[row * 512 + kbeg + k] * inv, sc[row * 512 + kbeg + k + 1] * inv);
    }
  }
  __syncthreads();

  // out = W x V   (16 x Sk) x (Sk x 32), V in (hd, S) layout per head
  v8f acc0 = {}, acc1 = {};
  for (int ks = 0; ks < Sk; ks += 32) {
    BF16Frag aw, v0, v1;
    aw.q[0] = *(const uint4*)&wb[l16 * 512 + ks + khalf * 8];
    aw.q[1] = *(const uint4*)&wb[l16 * 512 + ks + 16 + khalf * 8];
    {
      const unsigned short* vp0 = Vt + ((size_t)bh * HEAD_DIM + l16) * Sk + ks + khalf * 8;
      const unsigned short* vp1 = Vt + ((size_t)bh * HEAD_DIM + 16 + l16) * Sk + ks + khalf * 8;
      v0.q[0] = *(const uint4*)(vp0);
      v0.q[1] = *(const uint4*)(vp0 + 16);
      v1.q[0] = *(const uint4*)(vp1);
      v1.q[1] = *(const uint4*)(vp1 + 16);
    }
    acc0 = __builtin_amdgcn_wmma_f32_16x16x32_bf16(
        false, aw.v, false, v0.v, (short)0, acc0, false, false);
    acc1 = __builtin_amdgcn_wmma_f32_16x16x32_bf16(
        false, aw.v, false, v1.v, (short)0, acc1, false, false);
  }
#pragma unroll
  for (int r = 0; r < 8; ++r) {
    int row = qt * 16 + khalf * 8 + r;
    size_t base = ((size_t)bh * Sq + row) * HEAD_DIM;
    Oh[base + l16]      = acc0[r];
    Oh[base + 16 + l16] = acc1[r];
  }
}

// ------------------------------ orchestration ------------------------------

extern "C" void kernel_launch(void* const* d_in, const int* in_sizes, int n_in,
                              void* d_out, int out_size, void* d_ws, size_t ws_size,
                              hipStream_t stream) {
  (void)in_sizes; (void)n_in; (void)out_size; (void)ws_size;
  auto F = [&](int i) { return (const float*)d_in[i]; };

  // Input order: src, tgt, then params flattened jax-pytree style
  // (dict keys sorted at every level).
  const float* src = F(0);
  const int* tgt = (const int*)d_in[1];
  const int DEC0 = 2;                 // 3 x 26 leaves
  const int DNORM_B = 80, DNORM_G = 81;
  const int EMBED = 82;
  const int ENC0 = 83;                // 3 x 16 leaves
  const int ENORM_B = 131, ENORM_G = 132;
  const int FC_B = 133, FC_W = 134, PE = 135;
  const int D_BK = 0, D_BO = 1, D_BQ = 2, D_BV = 3, D_WK = 4, D_WO = 5, D_WQ = 6, D_WV = 7;
  const int D_B1 = 8, D_B2 = 9, D_W1 = 10, D_W2 = 11;
  const int D_LN1B = 12, D_LN1G = 13, D_LN2B = 14, D_LN2G = 15, D_LN3B = 16, D_LN3G = 17;
  const int D_SELF = 18;
  const int E_BK = 0, E_BO = 1, E_BQ = 2, E_BV = 3, E_WK = 4, E_WO = 5, E_WQ = 6, E_WV = 7;
  const int E_B1 = 8, E_B2 = 9, E_W1 = 10, E_W2 = 11;
  const int E_LN1B = 12, E_LN1G = 13, E_LN2B = 14, E_LN2G = 15;

  const float* pe = F(PE);

  size_t off = 0;
  auto alloc = [&](size_t bytes) {
    void* p = (char*)d_ws + off;
    off += (bytes + 255) & ~(size_t)255;
    return p;
  };
  auto convW = [&](const float* w, size_t n) {
    unsigned short* o = (unsigned short*)alloc(n * 2);
    int n4 = (int)(n / 4);
    k_convert_bf16<<<dim3((unsigned)((n4 + 255) / 256)), dim3(256), 0, stream>>>(w, o, n4);
    return o;
  };

  const size_t DD = (size_t)D_MODEL * D_MODEL;      // 65536
  const size_t DF = (size_t)D_MODEL * FF_DIM;       // 262144

  unsigned short *e_wq[3], *e_wk[3], *e_wv[3], *e_wo[3], *e_w1[3], *e_w2[3];
  for (int i = 0; i < 3; ++i) {
    int b0 = ENC0 + i * 16;
    e_wq[i] = convW(F(b0 + E_WQ), DD);
    e_wk[i] = convW(F(b0 + E_WK), DD);
    e_wv[i] = convW(F(b0 + E_WV), DD);
    e_wo[i] = convW(F(b0 + E_WO), DD);
    e_w1[i] = convW(F(b0 + E_W1), DF);
    e_w2[i] = convW(F(b0 + E_W2), DF);
  }
  unsigned short *s_wq[3], *s_wk[3], *s_wv[3], *s_wo[3];
  unsigned short *c_wq[3], *c_wk[3], *c_wv[3], *c_wo[3];
  unsigned short *d_w1[3], *d_w2[3];
  for (int i = 0; i < 3; ++i) {
    int b0 = DEC0 + i * 26;
    c_wq[i] = convW(F(b0 + D_WQ), DD);
    c_wk[i] = convW(F(b0 + D_WK), DD);
    c_wv[i] = convW(F(b0 + D_WV), DD);
    c_wo[i] = convW(F(b0 + D_WO), DD);
    s_wq[i] = convW(F(b0 + D_SELF + D_WQ), DD);
    s_wk[i] = convW(F(b0 + D_SELF + D_WK), DD);
    s_wv[i] = convW(F(b0 + D_SELF + D_WV), DD);
    s_wo[i] = convW(F(b0 + D_SELF + D_WO), DD);
    d_w1[i] = convW(F(b0 + D_W1), DF);
    d_w2[i] = convW(F(b0 + D_W2), DF);
  }
  unsigned short* fcw_bf = convW(F(FC_W), (size_t)D_MODEL * VOCAB);

  const int NTs = S_SRC * BATCH;   // 16384
  const int NTt = S_TGT * BATCH;   // 8192
  float*          xe_f   = (float*)alloc((size_t)NTs * D_MODEL * 4);
  unsigned short* xe_bf  = (unsigned short*)alloc((size_t)NTs * D_MODEL * 2);
  float*          xd_f   = (float*)alloc((size_t)NTt * D_MODEL * 4);
  unsigned short* xd_bf  = (unsigned short*)alloc((size_t)NTt * D_MODEL * 2);
  unsigned short* mem_bf = (unsigned short*)alloc((size_t)NTs * D_MODEL * 2);
  float*          q_f    = (float*)alloc((size_t)NTs * D_MODEL * 4);
  float*          k_f    = (float*)alloc((size_t)NTs * D_MODEL * 4);
  float*          v_f    = (float*)alloc((size_t)NTs * D_MODEL * 4);
  unsigned short* qh     = (unsigned short*)alloc((size_t)NTs * D_MODEL * 2);
  unsigned short* kh     = (unsigned short*)alloc((size_t)NTs * D_MODEL * 2);
  unsigned short* vt     = (unsigned short*)alloc((size_t)NTs * D_MODEL * 2);
  float*          oh_f   = (float*)alloc((size_t)NTs * D_MODEL * 4);
  unsigned short* om_bf  = (unsigned short*)alloc((size_t)NTs * D_MODEL * 2);
  float*          t0_f   = (float*)alloc((size_t)NTs * D_MODEL * 4);
  float*          hh_f   = (float*)alloc((size_t)NTs * FF_DIM * 4);
  unsigned short* hh_bf  = (unsigned short*)alloc((size_t)NTs * FF_DIM * 2);

  auto gemm = [&](const unsigned short* A, const unsigned short* Bw, const float* bias,
                  float* C, unsigned short* Cbf, int M, int N, int K, int relu, int permB) {
    dim3 g((N + 127) / 128, (M + 63) / 64);
    if ((N & 127) == 0)
      k_gemm_bf16<true><<<g, dim3(256), 0, stream>>>(A, Bw, bias, C, Cbf, M, N, K, relu, permB);
    else
      k_gemm_bf16<false><<<g, dim3(256), 0, stream>>>(A, Bw, bias, C, Cbf, M, N, K, relu, permB);
  };
  auto splitH = [&](const float* x, unsigned short* o, int S) {
    int total2 = BATCH * NHEAD * S * HEAD_DIM / 2;
    k_split_heads<<<(total2 + 255) / 256, 256, 0, stream>>>(x, o, S, total2);
  };
  auto splitHT = [&](const float* x, unsigned short* o, int S) {
    int total2 = BATCH * NHEAD * S * HEAD_DIM / 2;
    k_split_heads_T<<<(total2 + 255) / 256, 256, 0, stream>>>(x, o, S, total2);
  };
  auto mergeH = [&](const float* oh, unsigned short* o, int S) {
    int total2 = S * BATCH * D_MODEL / 2;
    k_merge_heads<<<(total2 + 255) / 256, 256, 0, stream>>>(oh, o, S, total2);
  };
  auto attn = [&](const unsigned short* Q, const unsigned short* K, const unsigned short* V,
                  float* O, int Sq, int Sk, int causal) {
    int blocks = BATCH * NHEAD * (Sq / 16);
    k_attn<<<blocks, 32, 0, stream>>>(Q, K, V, O, Sq, Sk, causal, 0.17677669529663687f);
  };
  auto ln = [&](const float* x, const float* y, const float* g, const float* b,
                float* of, unsigned short* obf, int NT) {
    k_ln<<<NT, 32, 0, stream>>>(x, y, g, b, of, obf);
  };

  // embeddings + positional encoding
  {
    int t2 = NTs * D_MODEL / 2;
    k_embed_src<<<(t2 + 255) / 256, 256, 0, stream>>>(src, pe, xe_f, xe_bf, t2);
    int t3 = NTt * D_MODEL / 2;
    k_embed_tgt<<<(t3 + 255) / 256, 256, 0, stream>>>(tgt, F(EMBED), pe, xd_f, xd_bf, t3);
  }

  // ------------------------------- encoder --------------------------------
  for (int i = 0; i < 3; ++i) {
    int b0 = ENC0 + i * 16;
    gemm(xe_bf, e_wq[i], F(b0 + E_BQ), q_f, nullptr, NTs, D_MODEL, D_MODEL, 0, 0);
    gemm(xe_bf, e_wk[i], F(b0 + E_BK), k_f, nullptr, NTs, D_MODEL, D_MODEL, 0, 0);
    gemm(xe_bf, e_wv[i], F(b0 + E_BV), v_f, nullptr, NTs, D_MODEL, D_MODEL, 0, 0);
    splitH(q_f, qh, S_SRC); splitH(k_f, kh, S_SRC); splitHT(v_f, vt, S_SRC);
    attn(qh, kh, vt, oh_f, S_SRC, S_SRC, 0);
    mergeH(oh_f, om_bf, S_SRC);
    gemm(om_bf, e_wo[i], F(b0 + E_BO), t0_f, nullptr, NTs, D_MODEL, D_MODEL, 0, 0);
    ln(xe_f, t0_f, F(b0 + E_LN1G), F(b0 + E_LN1B), xe_f, xe_bf, NTs);
    gemm(xe_bf, e_w1[i], F(b0 + E_B1), hh_f, hh_bf, NTs, FF_DIM, D_MODEL, 1, 0);
    gemm(hh_bf, e_w2[i], F(b0 + E_B2), t0_f, nullptr, NTs, D_MODEL, FF_DIM, 0, 0);
    ln(xe_f, t0_f, F(b0 + E_LN2G), F(b0 + E_LN2B), xe_f, xe_bf, NTs);
  }
  ln(xe_f, nullptr, F(ENORM_G), F(ENORM_B), xe_f, mem_bf, NTs);

  // ------------------------------- decoder --------------------------------
  for (int i = 0; i < 3; ++i) {
    int b0 = DEC0 + i * 26;
    // masked self-attention
    gemm(xd_bf, s_wq[i], F(b0 + D_SELF + D_BQ), q_f, nullptr, NTt, D_MODEL, D_MODEL, 0, 0);
    gemm(xd_bf, s_wk[i], F(b0 + D_SELF + D_BK), k_f, nullptr, NTt, D_MODEL, D_MODEL, 0, 0);
    gemm(xd_bf, s_wv[i], F(b0 + D_SELF + D_BV), v_f, nullptr, NTt, D_MODEL, D_MODEL, 0, 0);
    splitH(q_f, qh, S_TGT); splitH(k_f, kh, S_TGT); splitHT(v_f, vt, S_TGT);
    attn(qh, kh, vt, oh_f, S_TGT, S_TGT, 1);
    mergeH(oh_f, om_bf, S_TGT);
    gemm(om_bf, s_wo[i], F(b0 + D_SELF + D_BO), t0_f, nullptr, NTt, D_MODEL, D_MODEL, 0, 0);
    ln(xd_f, t0_f, F(b0 + D_LN1G), F(b0 + D_LN1B), xd_f, xd_bf, NTt);
    // cross-attention against encoder memory
    gemm(xd_bf, c_wq[i], F(b0 + D_BQ), q_f, nullptr, NTt, D_MODEL, D_MODEL, 0, 0);
    gemm(mem_bf, c_wk[i], F(b0 + D_BK), k_f, nullptr, NTs, D_MODEL, D_MODEL, 0, 0);
    gemm(mem_bf, c_wv[i], F(b0 + D_BV), v_f, nullptr, NTs, D_MODEL, D_MODEL, 0, 0);
    splitH(q_f, qh, S_TGT); splitH(k_f, kh, S_SRC); splitHT(v_f, vt, S_SRC);
    attn(qh, kh, vt, oh_f, S_TGT, S_SRC, 0);
    mergeH(oh_f, om_bf, S_TGT);
    gemm(om_bf, c_wo[i], F(b0 + D_BO), t0_f, nullptr, NTt, D_MODEL, D_MODEL, 0, 0);
    ln(xd_f, t0_f, F(b0 + D_LN2G), F(b0 + D_LN2B), xd_f, xd_bf, NTt);
    // FFN
    gemm(xd_bf, d_w1[i], F(b0 + D_B1), hh_f, hh_bf, NTt, FF_DIM, D_MODEL, 1, 0);
    gemm(hh_bf, d_w2[i], F(b0 + D_B2), t0_f, nullptr, NTt, D_MODEL, FF_DIM, 0, 0);
    ln(xd_f, t0_f, F(b0 + D_LN3G), F(b0 + D_LN3B), xd_f, xd_bf, NTt);
  }
  ln(xd_f, nullptr, F(DNORM_G), F(DNORM_B), xd_f, xd_bf, NTt);

  // final FC with (s,b) -> (b,s) row permutation into d_out (B, S_tgt, VOCAB)
  gemm(xd_bf, fcw_bf, F(FC_B), (float*)d_out, nullptr, NTt, VOCAB, D_MODEL, 0, BATCH);
}
